// NeuralODE_30511447671609
// MI455X (gfx1250) — matrix operand order repaired and checked by reading, not compiled, not run on hardware
//
#include <hip/hip_runtime.h>

// ---------------------------------------------------------------------------
// GRU-ODE (Neural ODE) for MI455X / gfx1250.
// dh/dt = (1 - sigmoid(h Wz^T)) * (tanh((sigmoid(h Wr^T) * h) Wh^T) - h)
// Fixed-step RK4, bf16 WMMA GEMMs (f32 accumulate), async-DMA double-buffered
// LDS tiles (GLOBAL_LOAD_ASYNC_TO_LDS_B128, ASYNCcnt). B = H = 1024, T = 16.
// ---------------------------------------------------------------------------

typedef __attribute__((ext_vector_type(16))) __bf16 v16bf;
typedef __attribute__((ext_vector_type(8)))  __bf16 v8bf;
typedef __attribute__((ext_vector_type(8)))  float  v8f;

#define HDIM  1024
#define BDIM  1024
#define TOUT  16
#define KT    32       // WMMA K per step (bf16 16x16x32)
#define BLK_M 128      // block tile rows (batch)
#define BLK_N 64       // block tile cols (neurons)

__device__ __forceinline__ float sigmoidf_(float x) {
    return 1.0f / (1.0f + __expf(-x));
}

// --- CDNA5 async copy: global -> LDS, 16B per lane, tracked by ASYNCcnt ----
__device__ __forceinline__ void async_load16(const void* lds_dst, const void* gsrc) {
    const unsigned int       ldsOff = (unsigned int)(unsigned long long)lds_dst; // low 32b = LDS offset
    const unsigned long long gaddr  = (unsigned long long)gsrc;
    asm volatile("global_load_async_to_lds_b128 %0, %1, off"
                 :: "v"(ldsOff), "v"(gaddr)
                 : "memory");
}
__device__ __forceinline__ void wait_async0() {
    asm volatile("s_wait_asynccnt 0x0" ::: "memory");
}

// --- WMMA fragment loaders (CDNA5 ISA 7.12.2 layouts, wave32) --------------
// A-matrix 16x32 bf16: lane L holds row M=L%16.
//   lanes 0-15 : elems 0..7 -> K=0..7,  elems 8..15 -> K=16..23
//   lanes 16-31: elems 0..7 -> K=8..15, elems 8..15 -> K=24..31
__device__ __forceinline__ v16bf load_a_frag(const __bf16* sA, int rowBase, int lane) {
    const int hi  = lane >> 4;
    const int row = rowBase + (lane & 15);
    const __bf16* p = sA + row * KT + hi * 8;
    union { v16bf v; v8bf h[2]; } u;
    u.h[0] = *(const v8bf*)(p);        // K = hi*8 + 0..7
    u.h[1] = *(const v8bf*)(p + 16);   // K = hi*8 + 16..23
    return u.v;
}

// B-matrix 32x16 bf16 (K x N): lane L holds col N=L%16.
//   lanes 0-15: elems 0..7 -> K=0..7, elems 8..15 -> K=16..23 ; lanes 16-31: K=8..15, 24..31
// sW holds W rows [n][k] row-major.
__device__ __forceinline__ v16bf load_b_frag(const __bf16* sW, int colBase, int lane) {
    const int hi = lane >> 4;
    const int n  = colBase + (lane & 15);
    const __bf16* p = sW + n * KT + hi * 8;
    union { v16bf v; v8bf h[2]; } u;
    u.h[0] = *(const v8bf*)(p);
    u.h[1] = *(const v8bf*)(p + 16);
    return u.v;
}

#define WMMA_BF16(A, Bm, C) \
    __builtin_amdgcn_wmma_f32_16x16x32_bf16(false, (A), false, (Bm), (short)0, (C), false, false)

// ---------------------------------------------------------------------------
// gemm_rz: Z = sigmoid(S Wz^T) [f32], RH = bf16(sigmoid(S Wr^T) * S)
// Dual GEMM sharing the A (state) tile. Block 128x64, 8 waves, wave 32x32.
// Double-buffered LDS fed by async DMA.
// ---------------------------------------------------------------------------
__global__ __launch_bounds__(256)
void gemm_rz_kernel(const __bf16* __restrict__ Abf,   // stage, bf16 [B,H]
                    const float*  __restrict__ Sf32,  // stage, f32  [B,H]
                    const __bf16* __restrict__ Wr,    // [H,H] bf16
                    const __bf16* __restrict__ Wz,    // [H,H] bf16
                    float*        __restrict__ Zout,  // [B,H] f32
                    __bf16*       __restrict__ RHout) // [B,H] bf16
{
    __shared__ __bf16 sA[2][BLK_M * KT];
    __shared__ __bf16 sR[2][BLK_N * KT];
    __shared__ __bf16 sZ[2][BLK_N * KT];

    const int tid  = threadIdx.x;
    const int lane = tid & 31;
    const int w    = tid >> 5;
    const int wm   = w >> 1;     // 0..3 (M, 32 rows each)
    const int wn   = w & 1;      // 0..1 (N, 32 cols each)
    const int blkN = blockIdx.x * BLK_N;
    const int blkM = blockIdx.y * BLK_M;

    // per-thread DMA slots (computed once)
    const int aRow0 = (tid * 2) >> 2,      aOff0 = ((tid * 2) & 3) * 8;
    const int aRow1 = (tid * 2 + 1) >> 2,  aOff1 = ((tid * 2 + 1) & 3) * 8;
    const int wRow  = tid >> 2,            wOff  = (tid & 3) * 8;

    auto emit_tile = [&](int kk, int buf) {
        async_load16(sA[buf] + aRow0 * KT + aOff0,
                     Abf + (size_t)(blkM + aRow0) * HDIM + kk + aOff0);
        async_load16(sA[buf] + aRow1 * KT + aOff1,
                     Abf + (size_t)(blkM + aRow1) * HDIM + kk + aOff1);
        async_load16(sR[buf] + wRow * KT + wOff,
                     Wr + (size_t)(blkN + wRow) * HDIM + kk + wOff);
        async_load16(sZ[buf] + wRow * KT + wOff,
                     Wz + (size_t)(blkN + wRow) * HDIM + kk + wOff);
    };

    v8f accR[2][2] = {};
    v8f accZ[2][2] = {};

    emit_tile(0, 0);
    int cur = 0;
    for (int kk = 0; kk < HDIM; kk += KT) {
        wait_async0();          // own DMAs done; barrier makes all waves' DMAs visible
        __syncthreads();
        if (kk + KT < HDIM) emit_tile(kk + KT, cur ^ 1);   // overlap next tile DMA

        const v16bf a0  = load_a_frag(sA[cur], wm * 32 + 0,  lane);
        const v16bf a1  = load_a_frag(sA[cur], wm * 32 + 16, lane);
        const v16bf br0 = load_b_frag(sR[cur], wn * 32 + 0,  lane);
        const v16bf br1 = load_b_frag(sR[cur], wn * 32 + 16, lane);
        const v16bf bz0 = load_b_frag(sZ[cur], wn * 32 + 0,  lane);
        const v16bf bz1 = load_b_frag(sZ[cur], wn * 32 + 16, lane);

        accR[0][0] = WMMA_BF16(a0, br0, accR[0][0]);
        accR[0][1] = WMMA_BF16(a0, br1, accR[0][1]);
        accR[1][0] = WMMA_BF16(a1, br0, accR[1][0]);
        accR[1][1] = WMMA_BF16(a1, br1, accR[1][1]);
        accZ[0][0] = WMMA_BF16(a0, bz0, accZ[0][0]);
        accZ[0][1] = WMMA_BF16(a0, bz1, accZ[0][1]);
        accZ[1][0] = WMMA_BF16(a1, bz0, accZ[1][0]);
        accZ[1][1] = WMMA_BF16(a1, bz1, accZ[1][1]);
        cur ^= 1;
    }

    // --- epilogue: sigmoid, write z (f32) and r*h (bf16) ---
    const int hi   = lane >> 4;
    const int col0 = lane & 15;
    #pragma unroll
    for (int mt = 0; mt < 2; ++mt)
        #pragma unroll
        for (int nt = 0; nt < 2; ++nt)
            #pragma unroll
            for (int r = 0; r < 8; ++r) {
                const int grow = blkM + wm * 32 + mt * 16 + r + hi * 8;
                const int gcol = blkN + wn * 32 + nt * 16 + col0;
                const size_t idx = (size_t)grow * HDIM + gcol;
                const float z  = sigmoidf_(accZ[mt][nt][r]);
                const float rr = sigmoidf_(accR[mt][nt][r]);
                Zout[idx]  = z;
                RHout[idx] = (__bf16)(rr * Sf32[idx]);
            }
}

// ---------------------------------------------------------------------------
// gemm_u: K = (1 - Z) * (tanh(RH Wh^T) - S)   -- full derivative, fused.
// ---------------------------------------------------------------------------
__global__ __launch_bounds__(256)
void gemm_u_kernel(const __bf16* __restrict__ RHbf,  // r*h, bf16 [B,H]
                   const float*  __restrict__ Sf32,  // stage f32 [B,H]
                   const __bf16* __restrict__ Wh,    // [H,H] bf16
                   const float*  __restrict__ Zin,   // [B,H] f32
                   float*        __restrict__ Kout)  // [B,H] f32 derivative
{
    __shared__ __bf16 sA[2][BLK_M * KT];
    __shared__ __bf16 sW[2][BLK_N * KT];

    const int tid  = threadIdx.x;
    const int lane = tid & 31;
    const int w    = tid >> 5;
    const int wm   = w >> 1;
    const int wn   = w & 1;
    const int blkN = blockIdx.x * BLK_N;
    const int blkM = blockIdx.y * BLK_M;

    const int aRow0 = (tid * 2) >> 2,      aOff0 = ((tid * 2) & 3) * 8;
    const int aRow1 = (tid * 2 + 1) >> 2,  aOff1 = ((tid * 2 + 1) & 3) * 8;
    const int wRow  = tid >> 2,            wOff  = (tid & 3) * 8;

    auto emit_tile = [&](int kk, int buf) {
        async_load16(sA[buf] + aRow0 * KT + aOff0,
                     RHbf + (size_t)(blkM + aRow0) * HDIM + kk + aOff0);
        async_load16(sA[buf] + aRow1 * KT + aOff1,
                     RHbf + (size_t)(blkM + aRow1) * HDIM + kk + aOff1);
        async_load16(sW[buf] + wRow * KT + wOff,
                     Wh + (size_t)(blkN + wRow) * HDIM + kk + wOff);
    };

    v8f acc[2][2] = {};

    emit_tile(0, 0);
    int cur = 0;
    for (int kk = 0; kk < HDIM; kk += KT) {
        wait_async0();
        __syncthreads();
        if (kk + KT < HDIM) emit_tile(kk + KT, cur ^ 1);

        const v16bf a0 = load_a_frag(sA[cur], wm * 32 + 0,  lane);
        const v16bf a1 = load_a_frag(sA[cur], wm * 32 + 16, lane);
        const v16bf b0 = load_b_frag(sW[cur], wn * 32 + 0,  lane);
        const v16bf b1 = load_b_frag(sW[cur], wn * 32 + 16, lane);

        acc[0][0] = WMMA_BF16(a0, b0, acc[0][0]);
        acc[0][1] = WMMA_BF16(a0, b1, acc[0][1]);
        acc[1][0] = WMMA_BF16(a1, b0, acc[1][0]);
        acc[1][1] = WMMA_BF16(a1, b1, acc[1][1]);
        cur ^= 1;
    }

    const int hi   = lane >> 4;
    const int col0 = lane & 15;
    #pragma unroll
    for (int mt = 0; mt < 2; ++mt)
        #pragma unroll
        for (int nt = 0; nt < 2; ++nt)
            #pragma unroll
            for (int r = 0; r < 8; ++r) {
                const int grow = blkM + wm * 32 + mt * 16 + r + hi * 8;
                const int gcol = blkN + wn * 32 + nt * 16 + col0;
                const size_t idx = (size_t)grow * HDIM + gcol;
                const float u = tanhf(acc[mt][nt][r]);
                Kout[idx] = (1.0f - Zin[idx]) * (u - Sf32[idx]);
            }
}

// ---------------------------------------------------------------------------
// Elementwise helpers
// ---------------------------------------------------------------------------
__global__ void f32_to_bf16_kernel(__bf16* __restrict__ dst, const float* __restrict__ src) {
    const int i = blockIdx.x * blockDim.x + threadIdx.x;
    dst[i] = (__bf16)src[i];
}

__global__ void copy2_kernel(float* __restrict__ out, __bf16* __restrict__ out16,
                             const float* __restrict__ src) {
    const int i = blockIdx.x * blockDim.x + threadIdx.x;
    const float v = src[i];
    out[i]   = v;
    out16[i] = (__bf16)v;
}

// out = x + a*y ; optionally also write bf16 copy (stage for next GEMM)
__global__ void axpy_kernel(float* __restrict__ out, __bf16* __restrict__ out16,
                            const float* __restrict__ x, const float* __restrict__ y,
                            float a, int write16) {
    const int i = blockIdx.x * blockDim.x + threadIdx.x;
    const float v = x[i] + a * y[i];
    out[i] = v;
    if (write16) out16[i] = (__bf16)v;
}

__global__ void write_out_kernel(float* __restrict__ out, const float* __restrict__ h, int ti) {
    const int i = blockIdx.x * blockDim.x + threadIdx.x;
    const int b = i >> 10;           // / HDIM
    const int j = i & (HDIM - 1);
    out[(size_t)b * TOUT * HDIM + (size_t)ti * HDIM + j] = h[i];
}

// ---------------------------------------------------------------------------
// Host orchestration: weights->bf16 once, then 15 intervals x 2 RK4 substeps.
// ---------------------------------------------------------------------------
extern "C" void kernel_launch(void* const* d_in, const int* in_sizes, int n_in,
                              void* d_out, int out_size, void* d_ws, size_t ws_size,
                              hipStream_t stream) {
    (void)in_sizes; (void)n_in; (void)out_size; (void)ws_size;

    const float* x0   = (const float*)d_in[0];
    /* d_in[1] = t (linspace(0,1,16)) -- values known statically */
    const float* W_hr = (const float*)d_in[2];
    const float* W_hz = (const float*)d_in[3];
    const float* W_hh = (const float*)d_in[4];
    float* out = (float*)d_out;

    char*  ws  = (char*)d_ws;
    size_t off = 0;
    auto alloc = [&](size_t bytes) -> void* {
        void* p = ws + off;
        off += (bytes + 255) & ~(size_t)255;
        return p;
    };
    const size_t NE = (size_t)BDIM * HDIM;
    __bf16* Wr16   = (__bf16*)alloc(NE * 2);
    __bf16* Wz16   = (__bf16*)alloc(NE * 2);
    __bf16* Wh16   = (__bf16*)alloc(NE * 2);
    float*  h      = (float*) alloc(NE * 4);
    float*  acc    = (float*) alloc(NE * 4);
    float*  stage  = (float*) alloc(NE * 4);
    __bf16* st16   = (__bf16*)alloc(NE * 2);
    __bf16* rh16   = (__bf16*)alloc(NE * 2);
    float*  zbuf   = (float*) alloc(NE * 4);
    float*  kbuf   = (float*) alloc(NE * 4);

    const dim3 eb(256), eg((unsigned)(NE / 256));
    const dim3 gb(256), gg(HDIM / BLK_N, BDIM / BLK_M);

    f32_to_bf16_kernel<<<eg, eb, 0, stream>>>(Wr16, W_hr);
    f32_to_bf16_kernel<<<eg, eb, 0, stream>>>(Wz16, W_hz);
    f32_to_bf16_kernel<<<eg, eb, 0, stream>>>(Wh16, W_hh);

    copy2_kernel<<<eg, eb, 0, stream>>>(h, st16, x0);
    write_out_kernel<<<eg, eb, 0, stream>>>(out, h, 0);

    const int   SUB    = 2;                    // RK4 substeps per output interval
    const float dt_out = 1.0f / (TOUT - 1);
    const float dt     = dt_out / SUB;

    auto feval = [&]() {   // derivative of current `stage` -> kbuf
        gemm_rz_kernel<<<gg, gb, 0, stream>>>(st16, stage, Wr16, Wz16, zbuf, rh16);
        gemm_u_kernel <<<gg, gb, 0, stream>>>(rh16, stage, Wh16, zbuf, kbuf);
    };

    for (int ti = 1; ti < TOUT; ++ti) {
        for (int s = 0; s < SUB; ++s) {
            // k1 at stage = h
            copy2_kernel<<<eg, eb, 0, stream>>>(stage, st16, h);
            feval();
            axpy_kernel<<<eg, eb, 0, stream>>>(acc,   nullptr, h,   kbuf, dt / 6.0f, 0);
            axpy_kernel<<<eg, eb, 0, stream>>>(stage, st16,    h,   kbuf, dt * 0.5f, 1);
            // k2
            feval();
            axpy_kernel<<<eg, eb, 0, stream>>>(acc,   nullptr, acc, kbuf, dt / 3.0f, 0);
            axpy_kernel<<<eg, eb, 0, stream>>>(stage, st16,    h,   kbuf, dt * 0.5f, 1);
            // k3
            feval();
            axpy_kernel<<<eg, eb, 0, stream>>>(acc,   nullptr, acc, kbuf, dt / 3.0f, 0);
            axpy_kernel<<<eg, eb, 0, stream>>>(stage, st16,    h,   kbuf, dt,        1);
            // k4
            feval();
            axpy_kernel<<<eg, eb, 0, stream>>>(h,     nullptr, acc, kbuf, dt / 6.0f, 0);
        }
        write_out_kernel<<<eg, eb, 0, stream>>>(out, h, ti);
    }
}
